// _DKGModule_74448963108922
// MI455X (gfx1250) — matrix operand order repaired and checked by reading, not compiled, not run on hardware
//
#include <hip/hip_runtime.h>
#include <hip/hip_bf16.h>

// MI455X / gfx1250, wave32. bf16 WMMA (v_wmma_f32_16x16x32_bf16) for the two
// GEMM-shaped ops; f32 VALU for the cheap dynamic depthwise path.
// k_stat: halo-row implicit GEMM over a pre-converted bf16 x2 copy -- each
// source row staged once per (dh,cblk) with +/-1 pixel halo; the three dw taps
// reuse it via shifted LDS reads. Double-buffered LDS pipeline, wave-uniform
// boundary branches, slim register staging (no f32->bf16 in the hot loop).

typedef __attribute__((ext_vector_type(16))) __bf16 v16bf;
typedef __attribute__((ext_vector_type(8)))  __bf16 v8bf;
typedef __attribute__((ext_vector_type(8)))  float  v8f;

#define NIMG 32
#define CTOT 512
#define CH   256
#define HH   48
#define WW   48
#define HW   2304
#define SA   40   // LDS row stride (bf16 elems) for A tiles [64][SA]
#define SB   40   // LDS row stride for B tile rows [w-position][SB]

static __device__ __forceinline__ __bf16 f2bf(float f) {
  unsigned u = __float_as_uint(f);
  unsigned r = (u + 0x7FFFu + ((u >> 16) & 1u)) >> 16;
  unsigned short s = (unsigned short)r;
  return __builtin_bit_cast(__bf16, s);
}

// ---------------- weight packing ----------------
__global__ void k_pack_fuse_w(const float* __restrict__ fw, __bf16* __restrict__ fwt) {
  int i = blockIdx.x * 256 + threadIdx.x;            // 512*512 elems, [o][c]
  fwt[i] = f2bf(fw[i]);
}

__global__ void k_pack_stat_w(const float* __restrict__ sw, __bf16* __restrict__ wtt) {
  int i = blockIdx.x * 256 + threadIdx.x;            // 9*256*256 elems
  int j = i / (256 * 256);                           // tap index
  int rem = i - j * 256 * 256;
  int o = rem >> 8;
  int c = rem & 255;
  // wtt layout: [j][o][c] (contiguous c) -> A staging is pure vector copies
  wtt[i] = f2bf(sw[(o * 256 + c) * 9 + j]);
}

// ---------------- g reduction + dynamic taps ----------------
__global__ void k_taps(const float* __restrict__ x,
                       const float* __restrict__ kw, const float* __restrict__ kb,
                       const float* __restrict__ ckw, const float* __restrict__ ckb,
                       float* __restrict__ taps) {
  int nc = blockIdx.x;                               // n*256 + c (c < 256 -> x1)
  int n = nc >> 8, c = nc & 255;
  const float* xp = x + ((size_t)(n * CTOT + c)) * HW;
  float s = 0.f;
  #pragma unroll
  for (int k = 0; k < 9; ++k) s += xp[threadIdx.x + k * 256];
  __shared__ float red[256];
  red[threadIdx.x] = s;
  __syncthreads();
  for (int st = 128; st > 0; st >>= 1) {
    if (threadIdx.x < st) red[threadIdx.x] += red[threadIdx.x + st];
    __syncthreads();
  }
  if (threadIdx.x == 0) {
    float g = red[0] * (1.0f / (float)HW);
    float a = kw[0], b = kb[0];
    #pragma unroll
    for (int j = 0; j < 9; ++j)
      taps[nc * 9 + j] = a * (g * ckw[j] + ckb[j]) + b;
  }
}

// ---- dynamic depthwise 3x3 -> cat[0..255], and x2 f32->bf16 conversion ----
__global__ void k_dyn_cvt(const float* __restrict__ x, const float* __restrict__ taps,
                          __bf16* __restrict__ catbf, __bf16* __restrict__ x2bf) {
  int nc = blockIdx.x;
  int n = nc >> 8, c = nc & 255;
  const float* xp  = x + ((size_t)(n * CTOT + c)) * HW;        // x1 channel
  const float* xp2 = x + ((size_t)(n * CTOT + CH + c)) * HW;   // x2 channel
  __bf16* op  = catbf + ((size_t)(n * CTOT + c)) * HW;
  __bf16* op2 = x2bf + ((size_t)(n * CH + c)) * HW;
  float t[9];
  #pragma unroll
  for (int j = 0; j < 9; ++j) t[j] = taps[nc * 9 + j];
  #pragma unroll
  for (int q = 0; q < 9; ++q) {
    int p = threadIdx.x + q * 256;
    int h = p / 48, w = p - h * 48;
    float acc = 0.f;
    #pragma unroll
    for (int j = 0; j < 9; ++j) {
      int hh = h + j / 3 - 1;
      int wc = w + j % 3 - 1;
      bool ok = ((unsigned)hh < 48u) && ((unsigned)wc < 48u);
      float v = xp[ok ? hh * 48 + wc : 0];     // clamped addr, unpredicated load
      acc += (ok ? v : 0.f) * t[j];
    }
    op[p]  = f2bf(acc);
    op2[p] = f2bf(xp2[p]);
  }
}

// ---------------- WMMA fragment builders (ISA wave32 bf16 layouts) ----------------
static __device__ __forceinline__ v16bf a_frag(const __bf16* As, int mt, int lane) {
  int m = lane & 15, h = lane >> 4;
  const __bf16* row = As + (mt * 16 + m) * SA;
  v8bf lo = *(const v8bf*)(row + h * 8);        // K = 8h .. 8h+7
  v8bf hi = *(const v8bf*)(row + 16 + h * 8);   // K = 16+8h .. 16+8h+7
  return __builtin_shufflevector(lo, hi, 0,1,2,3,4,5,6,7,8,9,10,11,12,13,14,15);
}
// B frag by LDS w-position row (pos = output col + dw for halo tiles)
static __device__ __forceinline__ v16bf b_frag(const __bf16* Bs, int pos0, int lane) {
  int nn = lane & 15, h = lane >> 4;
  const __bf16* col = Bs + (pos0 + nn) * SB;  // K = 8h..8h+7, 16+8h..16+8h+7
  v8bf lo = *(const v8bf*)(col + h * 8);
  v8bf hi = *(const v8bf*)(col + 16 + h * 8);
  return __builtin_shufflevector(lo, hi, 0,1,2,3,4,5,6,7,8,9,10,11,12,13,14,15);
}

// ---------------- static 3x3 conv, halo implicit GEMM -> cat[256..511] ----------------
// block: (n, row-pair, 64-out-channel tile), 256 threads = 8 waves (4 M x 2 rows)
// K loop: 3 dh x 8 channel-blocks; per iter: 3 A tiles (dw taps) + 1 halo'd B row,
// 9 WMMAs per wave from shifted LDS views.
__global__ void __launch_bounds__(256, 1)
k_stat(const __bf16* __restrict__ x2bf, const __bf16* __restrict__ wtt,
       const float* __restrict__ sb, __bf16* __restrict__ catbf) {
  const int n = blockIdx.x, rp = blockIdx.y, mb = blockIdx.z;
  const int tid = threadIdx.x, lane = tid & 31, wave = tid >> 5;
  const int mt = wave & 3, wrs = wave >> 2;            // wave's m-tile / row
  const int obase = mb * 64;
  __shared__ __bf16 As[2][3 * 64 * SA];                // [buf][tap][64][SA]
  __shared__ __bf16 Bs[2][2 * 50 * SB];                // [buf][row][w+1][SB]
  v8f acc0 = {}, acc1 = {}, acc2 = {};

  // staging thread mapping
  const int aoo = tid >> 2, akh = (tid & 3) * 8;       // A: v8bf per thread per tap
  const int kc = tid & 31, pg = tid >> 5;              // B: 12 pixels per thread
  const int prow = pg & 3, brs = pg >> 2;              // wave-uniform
  const int brow = rp * 2 + brs;

  // zero halo columns (w = -1 and w = 48) once; staging never touches them
  if (tid < 128) {
    int hk = tid & 31, q = tid >> 5;                   // q: row(2) x side(2)
    int pos = ((q >> 1) ? 49 : 0) + (q & 1) * 50;
    unsigned short z = 0;
    ((unsigned short*)&Bs[0][0])[pos * SB + hk] = z;
    ((unsigned short*)&Bs[1][0])[pos * SB + hk] = z;
  }

  v8bf ra[3];
  unsigned long long rb[3];                            // 12 contiguous bf16

  auto load_tile = [&](int kk) {
    const int dh = kk >> 3, cb = kk & 7;
    #pragma unroll
    for (int t = 0; t < 3; ++t)                        // 3 dw taps for this dh
      ra[t] = *(const v8bf*)(wtt + ((size_t)((dh * 3 + t) * 256 + obase + aoo)) * 256
                             + cb * 32 + akh);
    const int hh = brow + dh - 1;
    if ((unsigned)hh < 48u) {                          // wave-uniform row guard
      const __bf16* base = x2bf + ((size_t)(n * CH + cb * 32 + kc)) * HW
                           + hh * 48 + prow * 12;      // 8B aligned
      __builtin_prefetch(base + 32 * HW, 0, 1);        // next channel-block
      const unsigned long long* q = (const unsigned long long*)base;
      rb[0] = q[0]; rb[1] = q[1]; rb[2] = q[2];
    } else {
      rb[0] = 0; rb[1] = 0; rb[2] = 0;
    }
  };
  auto store_tile = [&](int buf) {
    #pragma unroll
    for (int t = 0; t < 3; ++t)
      *(v8bf*)(&As[buf][(t * 64 + aoo) * SA + akh]) = ra[t];
    unsigned short* bsu = (unsigned short*)&Bs[buf][0];
    #pragma unroll
    for (int it = 0; it < 12; ++it)                    // interior w-positions 1..48
      bsu[(brs * 50 + 1 + prow * 12 + it) * SB + kc] =
          (unsigned short)(rb[it >> 2] >> ((it & 3) * 16));
  };

  load_tile(0);
  store_tile(0);
  __syncthreads();
  for (int kk = 0; kk < 24; ++kk) {                    // 3 dh x 8 K-blocks of 32
    const int cur = kk & 1;
    if (kk + 1 < 24) load_tile(kk + 1);
    const __bf16* bsr = &Bs[cur][wrs * 50 * SB];
    #pragma unroll
    for (int t = 0; t < 3; ++t) {                      // dw = t-1 -> LDS pos pp+t
      v16bf a = a_frag(&As[cur][t * 64 * SA], mt, lane);
      v16bf b0 = b_frag(bsr, 0 + t, lane);
      v16bf b1 = b_frag(bsr, 16 + t, lane);
      v16bf b2 = b_frag(bsr, 32 + t, lane);
      acc0 = __builtin_amdgcn_wmma_f32_16x16x32_bf16(false, a, false, b0, (short)0, acc0, false, false);
      acc1 = __builtin_amdgcn_wmma_f32_16x16x32_bf16(false, a, false, b1, (short)0, acc1, false, false);
      acc2 = __builtin_amdgcn_wmma_f32_16x16x32_bf16(false, a, false, b2, (short)0, acc2, false, false);
    }
    if (kk + 1 < 24) store_tile((kk + 1) & 1);
    __syncthreads();
  }
  // epilogue: C/D layout -> lane col = lane&15, VGPR r holds m = r + 8*(lane>>4)
  const int row = rp * 2 + wrs;
  int nn = lane & 15, h = lane >> 4;
  #pragma unroll
  for (int r = 0; r < 8; ++r) {
    int o = obase + mt * 16 + r + 8 * h;
    float bias = sb[o];
    __bf16* dst = catbf + ((size_t)(n * CTOT + CH + o)) * HW + row * 48;
    dst[nn]      = f2bf(acc0[r] + bias);
    dst[16 + nn] = f2bf(acc1[r] + bias);
    dst[32 + nn] = f2bf(acc2[r] + bias);
  }
}

// ---------------- fuse 1x1 GEMM over cat (K=512) -> f32 out ----------------
__global__ void __launch_bounds__(256, 1)
k_fuse(const __bf16* __restrict__ catbf, const __bf16* __restrict__ fwt,
       const float* __restrict__ fb, float* __restrict__ out) {
  const int n = blockIdx.x, rp = blockIdx.y, mb = blockIdx.z;   // mb 0..7
  const int tid = threadIdx.x, lane = tid & 31, wave = tid >> 5;
  const int mt = wave & 3, wrs = wave >> 2;
  const int obase = mb * 64;
  __shared__ __bf16 As[2][64 * SA];
  __shared__ __bf16 Bs[2][96 * SB];
  v8f acc0 = {}, acc1 = {}, acc2 = {};

  const int aoo = tid >> 2, akh = (tid & 3) * 8;
  const int kc = tid & 31, pg = tid >> 5;
  const int prow = pg & 3, brs = pg >> 2;
  const int brow = rp * 2 + brs;

  v8bf ra;
  unsigned long long rb[3];                            // 12 contiguous bf16

  auto load_tile = [&](int cb) {
    ra = *(const v8bf*)(fwt + ((size_t)(obase + aoo)) * 512 + cb * 32 + akh);
    const __bf16* bp = catbf + ((size_t)(n * CTOT + cb * 32 + kc)) * HW
                       + brow * 48 + prow * 12;
    __builtin_prefetch(bp + 32 * HW, 0, 1);            // next K-block, same pixels
    const unsigned long long* q = (const unsigned long long*)bp;   // 8B aligned
    rb[0] = q[0]; rb[1] = q[1]; rb[2] = q[2];
  };
  auto store_tile = [&](int buf) {
    *(v8bf*)(&As[buf][aoo * SA + akh]) = ra;
    unsigned short* bsu = (unsigned short*)&Bs[buf][0];
    #pragma unroll
    for (int it = 0; it < 12; ++it)
      bsu[(brs * 48 + prow * 12 + it) * SB + kc] =
          (unsigned short)(rb[it >> 2] >> ((it & 3) * 16));
  };

  load_tile(0);
  store_tile(0);
  __syncthreads();
  for (int cb = 0; cb < 16; ++cb) {
    const int cur = cb & 1;
    if (cb + 1 < 16) load_tile(cb + 1);
    v16bf a  = a_frag(&As[cur][0], mt, lane);
    const __bf16* bsr = &Bs[cur][wrs * 48 * SB];
    v16bf b0 = b_frag(bsr, 0, lane);
    v16bf b1 = b_frag(bsr, 16, lane);
    v16bf b2 = b_frag(bsr, 32, lane);
    acc0 = __builtin_amdgcn_wmma_f32_16x16x32_bf16(false, a, false, b0, (short)0, acc0, false, false);
    acc1 = __builtin_amdgcn_wmma_f32_16x16x32_bf16(false, a, false, b1, (short)0, acc1, false, false);
    acc2 = __builtin_amdgcn_wmma_f32_16x16x32_bf16(false, a, false, b2, (short)0, acc2, false, false);
    if (cb + 1 < 16) store_tile((cb + 1) & 1);
    __syncthreads();
  }
  const int row = rp * 2 + wrs;
  int nn = lane & 15, h = lane >> 4;
  #pragma unroll
  for (int r = 0; r < 8; ++r) {
    int o = obase + mt * 16 + r + 8 * h;
    float bias = fb[o];
    float* dst = out + ((size_t)(n * CTOT + o)) * HW + row * 48;
    dst[nn]      = acc0[r] + bias;
    dst[16 + nn] = acc1[r] + bias;
    dst[32 + nn] = acc2[r] + bias;
  }
}

extern "C" void kernel_launch(void* const* d_in, const int* in_sizes, int n_in,
                              void* d_out, int out_size, void* d_ws, size_t ws_size,
                              hipStream_t stream) {
  (void)in_sizes; (void)n_in; (void)out_size; (void)ws_size;
  const float* x   = (const float*)d_in[0];
  const float* kw  = (const float*)d_in[1];
  const float* kb  = (const float*)d_in[2];
  const float* ckw = (const float*)d_in[3];
  const float* ckb = (const float*)d_in[4];
  const float* sw  = (const float*)d_in[5];
  const float* sb  = (const float*)d_in[6];
  const float* fw  = (const float*)d_in[7];
  const float* fb  = (const float*)d_in[8];
  float* out = (float*)d_out;

  // workspace carve-up (~116 MB total)
  char* ws = (char*)d_ws;
  size_t off = 0;
  float*  taps = (float*)(ws + off);  off += (size_t)NIMG * CH * 9 * 4;     off = (off + 255) & ~(size_t)255;
  __bf16* wtt  = (__bf16*)(ws + off); off += (size_t)9 * 256 * 256 * 2;     off = (off + 255) & ~(size_t)255;
  __bf16* fwt  = (__bf16*)(ws + off); off += (size_t)512 * 512 * 2;         off = (off + 255) & ~(size_t)255;
  __bf16* x2bf = (__bf16*)(ws + off); off += (size_t)NIMG * CH * HW * 2;    off = (off + 255) & ~(size_t)255;
  __bf16* catb = (__bf16*)(ws + off); off += (size_t)NIMG * CTOT * HW * 2;

  k_pack_fuse_w<<<1024, 256, 0, stream>>>(fw, fwt);
  k_pack_stat_w<<<2304, 256, 0, stream>>>(sw, wtt);
  k_taps<<<NIMG * CH, 256, 0, stream>>>(x, kw, kb, ckw, ckb, taps);
  k_dyn_cvt<<<NIMG * CH, 256, 0, stream>>>(x, taps, catb, x2bf);
  k_stat<<<dim3(NIMG, HH / 2, 4), 256, 0, stream>>>(x2bf, wtt, sb, catb);
  k_fuse<<<dim3(NIMG, HH / 2, 8), 256, 0, stream>>>(catb, fwt, fb, out);
}
